// WSAttention_61263413510506
// MI455X (gfx1250) — matrix operand order repaired and checked by reading, not compiled, not run on hardware
//
#include <hip/hip_runtime.h>
#include <hip/hip_bf16.h>
#include <stdint.h>

typedef _Float16 h16;
typedef __attribute__((ext_vector_type(16))) _Float16 v16h;
typedef __attribute__((ext_vector_type(8)))  _Float16 v8h;
typedef __attribute__((ext_vector_type(4)))  _Float16 v4h;
typedef __attribute__((ext_vector_type(2)))  _Float16 v2h;
typedef __attribute__((ext_vector_type(8)))  float    v8f;
typedef __attribute__((ext_vector_type(4)))  float    v4f;

#define B_   8
#define C_   64
#define H_   256
#define W_   256
#define HW_  (H_ * W_)
#define CS_  32
#define SCALE_ 0.17677669529663687f   // 32^-0.5

__device__ __forceinline__ v16h joinh(v8h lo, v8h hi) {
  union { v16h v; v8h h[2]; } u; u.h[0] = lo; u.h[1] = hi; return u.v;
}
__device__ __forceinline__ v8f wmma16(v16h a, v16h b, v8f c) {
  return __builtin_amdgcn_wmma_f32_16x16x32_f16(false, a, false, b, (short)0, c, false, false);
}
__device__ __forceinline__ v8h lds8(const h16* p) { return *reinterpret_cast<const v8h*>(p); }

// CDNA5 async global->LDS copies (VGLOBAL encoding, GV mode), tracked by ASYNCcnt.
__device__ __forceinline__ void async_ld_b64(unsigned lds_off, const void* g) {
  asm volatile("global_load_async_to_lds_b64 %0, %1, off"
               :: "v"(lds_off), "v"((unsigned long long)(uintptr_t)g) : "memory");
}
__device__ __forceinline__ void async_ld_b32(unsigned lds_off, const void* g) {
  asm volatile("global_load_async_to_lds_b32 %0, %1, off"
               :: "v"(lds_off), "v"((unsigned long long)(uintptr_t)g) : "memory");
}
__device__ __forceinline__ void wait_async0() {
  asm volatile("s_wait_asynccnt 0" ::: "memory");
}

// ---------------- K1: channel LayerNorm -> f16 ----------------
__global__ void k_ln(const float* __restrict__ x0, const float* __restrict__ lnw,
                     const float* __restrict__ lnb, h16* __restrict__ xh) {
  int pix = blockIdx.x * 256 + threadIdx.x;
  int b = pix / HW_, m = pix - b * HW_;
  const float* xp = x0 + (size_t)b * C_ * HW_ + m;
  float s = 0.f, s2 = 0.f;
  for (int c = 0; c < C_; ++c) { float v = xp[(size_t)c * HW_]; s += v; s2 += v * v; }
  float mu = s * (1.f / C_);
  float var = s2 * (1.f / C_) - mu * mu;
  float inv = rsqrtf(var + 1e-6f);
  h16* op = xh + (size_t)b * C_ * HW_ + m;
  for (int c = 0; c < C_; ++c) {
    float v = xp[(size_t)c * HW_];
    op[(size_t)c * HW_] = (h16)(((v - mu) * inv) * lnw[c] + lnb[c]);
  }
}

// ---------------- K2: depthwise 3x3 + ECA block-partial sums ----------------
__global__ void k_dw(const h16* __restrict__ xh, const float* __restrict__ dww,
                     const float* __restrict__ dwb, h16* __restrict__ x1,
                     float* __restrict__ partial) {
  int blk = blockIdx.x;
  int pb = blk & 255, bc = blk >> 8;
  int c = bc & (C_ - 1), b = bc >> 6;
  int m = pb * 256 + threadIdx.x;
  int y = m >> 8, x = m & 255;
  const h16* xp = xh + ((size_t)b * C_ + c) * HW_;
  float w[9];
#pragma unroll
  for (int i = 0; i < 9; ++i) w[i] = dww[c * 9 + i];
  float acc = dwb[c];
#pragma unroll
  for (int dy = -1; dy <= 1; ++dy)
#pragma unroll
    for (int dx = -1; dx <= 1; ++dx) {
      int yy = y + dy, xx = x + dx;
      if (yy >= 0 && yy < H_ && xx >= 0 && xx < W_)
        acc += w[(dy + 1) * 3 + (dx + 1)] * (float)xp[yy * W_ + xx];
    }
  x1[((size_t)b * C_ + c) * HW_ + m] = (h16)acc;
  __shared__ float red[256];
  red[threadIdx.x] = acc;
  __syncthreads();
  for (int s = 128; s > 0; s >>= 1) {
    if (threadIdx.x < s) red[threadIdx.x] += red[threadIdx.x + s];
    __syncthreads();
  }
  if (threadIdx.x == 0) partial[bc * 256 + pb] = red[0];
}

// ---------------- K3a: deterministic pool reduction ----------------
__global__ void k_pool(const float* __restrict__ partial, float* __restrict__ pool) {
  int bc = blockIdx.x * 256 + threadIdx.x;
  if (bc >= B_ * C_) return;
  float s = 0.f;
  for (int i = 0; i < 256; ++i) s += partial[bc * 256 + i];
  pool[bc] = s * (1.f / (float)HW_);
}

// ---------------- K3b: ECA channel attention (tiny) ----------------
__global__ void k_eca(const float* __restrict__ pool, const float* __restrict__ w1,
                      const float* __restrict__ w2, const float* __restrict__ w3,
                      const float* __restrict__ wc, float* __restrict__ ca) {
  __shared__ float mch[C_];
  int b = blockIdx.x, c = threadIdx.x;
  mch[c] = pool[b * C_ + c];
  __syncthreads();
  auto Y = [&](int j) -> float { return (j >= 0 && j < C_) ? mch[j] : 0.f; };
  auto Y1 = [&](int j) -> float {
    return (j >= 0 && j < C_) ? (w1[0] * Y(j - 1) + w1[1] * Y(j) + w1[2] * Y(j + 1)) : 0.f; };
  auto Y2 = [&](int j) -> float {
    return (j >= 0 && j < C_) ? (w2[0] * Y(j - 2) + w2[1] * Y(j) + w2[2] * Y(j + 2)) : 0.f; };
  auto Y3 = [&](int j) -> float {
    return (j >= 0 && j < C_) ? (w3[0] * Y(j - 4) + w3[1] * Y(j) + w3[2] * Y(j + 4)) : 0.f; };
  float o = 0.f;
#pragma unroll
  for (int k = 0; k < 3; ++k) {
    int j = c - 1 + k;
    o += wc[0 * 3 + k] * Y1(j) + wc[1 * 3 + k] * Y2(j) + wc[2 * 3 + k] * Y3(j);
  }
  ca[b * C_ + c] = 1.f / (1.f + __expf(-o));
}

// ---------------- K4: q/k/v 1x1 convs via WMMA, coalesced b128 output ---------
__global__ void k_qkv(const h16* __restrict__ xh, const float* __restrict__ wq,
                      const float* __restrict__ wk, const float* __restrict__ wv,
                      const float* __restrict__ ca, h16* __restrict__ q,
                      h16* __restrict__ k, h16* __restrict__ v) {
  __shared__ h16 Wl[3 * C_ * C_];
  __shared__ h16 Xt[128 * 72];          // reused as output staging after fragments built
  __shared__ float Cl[C_];
  int b = blockIdx.x / (HW_ / 128);
  int mb = (blockIdx.x % (HW_ / 128)) * 128;
  int tid = threadIdx.x;
  for (int i = tid; i < C_ * C_; i += 256) {
    Wl[i] = (h16)wq[i]; Wl[4096 + i] = (h16)wk[i]; Wl[8192 + i] = (h16)wv[i];
  }
  if (tid < C_) Cl[tid] = ca[b * C_ + tid];
  const h16* xp = xh + (size_t)b * C_ * HW_ + mb;
  for (int e = tid * 4; e < 128 * C_; e += 256 * 4) {
    int c = e >> 7, m = e & 127;
    v4h val = *reinterpret_cast<const v4h*>(xp + (size_t)c * HW_ + m);
#pragma unroll
    for (int j = 0; j < 4; ++j) Xt[(m + j) * 72 + c] = val[j];
  }
  __syncthreads();
  int w = tid >> 5, l = tid & 31, row = l & 15;
  bool hi = l >= 16;
  int pcol = w * 16;
  v16h bfrag[2];
#pragma unroll
  for (int ks = 0; ks < 2; ++ks) {
    int kb = ks * 32 + (hi ? 16 : 0);
    bfrag[ks] = joinh(lds8(&Xt[(pcol + row) * 72 + kb]), lds8(&Xt[(pcol + row) * 72 + kb + 8]));
  }
  h16* Ot = Xt;                          // safe to reuse after this barrier
  __syncthreads();
  for (int wi = 0; wi < 3; ++wi) {
    const h16* Wb = &Wl[wi * 4096];
    h16* outp = (wi == 0) ? q : (wi == 1) ? k : v;
    v8f acc[4];
#pragma unroll
    for (int rb = 0; rb < 4; ++rb) {
      acc[rb] = (v8f){0.f, 0.f, 0.f, 0.f, 0.f, 0.f, 0.f, 0.f};
#pragma unroll
      for (int ks = 0; ks < 2; ++ks) {
        int kb = ks * 32 + (hi ? 16 : 0);
        v16h a = joinh(lds8(&Wb[(rb * 16 + row) * 64 + kb]),
                       lds8(&Wb[(rb * 16 + row) * 64 + kb + 8]));
        acc[rb] = wmma16(a, bfrag[ks], acc[rb]);
      }
    }
#pragma unroll
    for (int rb = 0; rb < 4; ++rb)
#pragma unroll
      for (int r = 0; r < 8; ++r) {
        int o = rb * 16 + r + (hi ? 8 : 0);
        float val = acc[rb][r];
        if (wi == 2) val *= Cl[o];
        Ot[o * 128 + pcol + row] = (h16)val;
      }
    __syncthreads();
    for (int ch = tid; ch < 1024; ch += 256) {     // 64 o x 16 segs of 8 halves
      int o = ch >> 4, seg = ch & 15;
      v8h val = *reinterpret_cast<v8h*>(&Ot[o * 128 + seg * 8]);
      *reinterpret_cast<v8h*>(outp + ((size_t)b * C_ + o) * HW_ + mb + seg * 8) = val;
    }
    __syncthreads();
  }
}

// ---------------- K5: 2x2 stride-2 depthwise downsample of k,v ----------------
__global__ void k_down(const h16* __restrict__ kh, const h16* __restrict__ vh,
                       const float* __restrict__ dw, const float* __restrict__ db,
                       h16* __restrict__ kd, h16* __restrict__ vd) {
  int idx = blockIdx.x * 256 + threadIdx.x;
  int x = idx & 127, y = (idx >> 7) & 127, c = (idx >> 14) & 31, b = idx >> 19;
  const h16* kp = kh + ((size_t)b * C_ + 32 + c) * HW_;
  const h16* vp = vh + ((size_t)b * C_ + 32 + c) * HW_;
  float w0 = dw[c * 4], w1 = dw[c * 4 + 1], w2 = dw[c * 4 + 2], w3 = dw[c * 4 + 3];
  int p = (2 * y) * W_ + 2 * x;
  float kv = w0 * (float)kp[p] + w1 * (float)kp[p + 1] +
             w2 * (float)kp[p + W_] + w3 * (float)kp[p + W_ + 1] + db[c];
  float vv = w0 * (float)vp[p] + w1 * (float)vp[p + 1] +
             w2 * (float)vp[p + W_] + w3 * (float)vp[p + W_ + 1] + db[c];
  kd[idx] = (h16)kv;
  vd[idx] = (h16)vv;
}

// ---------------- K6: branch 0 window attention (ws=8) ------------------------
__global__ void k_attn0(const h16* __restrict__ q, const h16* __restrict__ k,
                        const h16* __restrict__ v, h16* __restrict__ y) {
  __shared__ h16 Ql[64 * 48], Kl[64 * 48], Vc[32 * 72], Al[64 * 72], Yl[32 * 72];
  int wid = blockIdx.x;
  int b = wid >> 10, wy = (wid >> 5) & 31, wx = wid & 31;
  int tid = threadIdx.x;
  bool wrapx = (wx == 31);
  size_t base = (size_t)b * C_ * HW_;
  // Q,K: 512 row-segments (mat, c, i); rows of 8 tokens, contiguous unless wrap.
  for (int sgi = tid; sgi < 512; sgi += 128) {
    int i = sgi & 7, c = (sgi >> 3) & 31, mat = sgi >> 8;
    int gy = (wy * 8 + i + 4) & 255;
    const h16* src = (mat ? k : q) + base + (size_t)c * HW_ + gy * W_;
    h16* dst = mat ? Kl : Ql;
    int t0 = i * 8;
    if (!wrapx) {
      int gx0 = wx * 8 + 4;
      v4h a = *reinterpret_cast<const v4h*>(src + gx0);
      v4h bb = *reinterpret_cast<const v4h*>(src + gx0 + 4);
#pragma unroll
      for (int j = 0; j < 4; ++j) { dst[(t0 + j) * 48 + c] = a[j]; dst[(t0 + 4 + j) * 48 + c] = bb[j]; }
    } else {
#pragma unroll
      for (int j = 0; j < 8; ++j) dst[(t0 + j) * 48 + c] = src[(wx * 8 + j + 4) & 255];
    }
  }
  // V: async global->LDS b64 copies (token-contiguous destination rows).
  if (!wrapx) {
    for (int sgi = tid; sgi < 512; sgi += 128) {
      int hf = sgi & 1, i = (sgi >> 1) & 7, c = sgi >> 4;
      int gy = (wy * 8 + i + 4) & 255, gx0 = wx * 8 + 4 + hf * 4;
      async_ld_b64((unsigned)(size_t)&Vc[c * 72 + i * 8 + hf * 4],
                   v + base + (size_t)c * HW_ + gy * W_ + gx0);
    }
  } else {
    for (int e = tid; e < 2048; e += 128) {
      int t = e & 63, c = e >> 6;
      int gy = (wy * 8 + (t >> 3) + 4) & 255, gx = (wx * 8 + (t & 7) + 4) & 255;
      Vc[c * 72 + t] = v[base + (size_t)c * HW_ + gy * W_ + gx];
    }
  }
  wait_async0();
  __syncthreads();
  int w = tid >> 5, l = tid & 31, row = l & 15;
  bool hi = l >= 16;
  int ab = hi ? 16 : 0;
  v16h aq = joinh(lds8(&Ql[(w * 16 + row) * 48 + ab]), lds8(&Ql[(w * 16 + row) * 48 + ab + 8]));
  v8f lg[4];
#pragma unroll
  for (int nt = 0; nt < 4; ++nt) {
    int kb = hi ? 16 : 0;
    v16h bk = joinh(lds8(&Kl[(nt * 16 + row) * 48 + kb]), lds8(&Kl[(nt * 16 + row) * 48 + kb + 8]));
    v8f z = {0.f, 0.f, 0.f, 0.f, 0.f, 0.f, 0.f, 0.f};
    lg[nt] = wmma16(aq, bk, z);
  }
#pragma unroll
  for (int nt = 0; nt < 4; ++nt)
#pragma unroll
    for (int r = 0; r < 8; ++r) lg[nt][r] *= SCALE_;
#pragma unroll
  for (int r = 0; r < 8; ++r) {
    float mx = fmaxf(fmaxf(lg[0][r], lg[1][r]), fmaxf(lg[2][r], lg[3][r]));
    for (int s = 1; s < 16; s <<= 1) mx = fmaxf(mx, __shfl_xor(mx, s, 32));
    float sm = 0.f;
#pragma unroll
    for (int nt = 0; nt < 4; ++nt) { float e = __expf(lg[nt][r] - mx); lg[nt][r] = e; sm += e; }
    for (int s = 1; s < 16; s <<= 1) sm += __shfl_xor(sm, s, 32);
    float inv = 1.f / sm;
#pragma unroll
    for (int nt = 0; nt < 4; ++nt) lg[nt][r] *= inv;
  }
#pragma unroll
  for (int nt = 0; nt < 4; ++nt)
#pragma unroll
    for (int r = 0; r < 8; ++r)
      Al[(w * 16 + r + (hi ? 8 : 0)) * 72 + nt * 16 + row] = (h16)lg[nt][r];
  __syncthreads();
  v8f oacc[2];
#pragma unroll
  for (int ct = 0; ct < 2; ++ct) {
    oacc[ct] = (v8f){0.f, 0.f, 0.f, 0.f, 0.f, 0.f, 0.f, 0.f};
#pragma unroll
    for (int ks = 0; ks < 2; ++ks) {
      int ak = ks * 32 + (hi ? 16 : 0);
      v16h aa = joinh(lds8(&Al[(w * 16 + row) * 72 + ak]),
                      lds8(&Al[(w * 16 + row) * 72 + ak + 8]));
      int vk = ks * 32 + (hi ? 16 : 0);
      v16h bb = joinh(lds8(&Vc[(ct * 16 + row) * 72 + vk]),
                      lds8(&Vc[(ct * 16 + row) * 72 + vk + 8]));
      oacc[ct] = wmma16(aa, bb, oacc[ct]);
    }
#pragma unroll
    for (int r = 0; r < 8; ++r)
      Yl[(ct * 16 + row) * 72 + w * 16 + r + (hi ? 8 : 0)] = (h16)oacc[ct][r];
  }
  __syncthreads();
  for (int sgi = tid; sgi < 256; sgi += 128) {       // 32 c x 8 rows
    int i = sgi & 7, c = sgi >> 3;
    int gy = (wy * 8 + i + 4) & 255;
    h16* dg = y + base + (size_t)c * HW_ + gy * W_;
    if (!wrapx) {
      int gx0 = wx * 8 + 4;
      *reinterpret_cast<v4h*>(dg + gx0)     = *reinterpret_cast<v4h*>(&Yl[c * 72 + i * 8]);
      *reinterpret_cast<v4h*>(dg + gx0 + 4) = *reinterpret_cast<v4h*>(&Yl[c * 72 + i * 8 + 4]);
    } else {
#pragma unroll
      for (int j = 0; j < 8; ++j) dg[(wx * 8 + j + 4) & 255] = Yl[c * 72 + i * 8 + j];
    }
  }
}

// ---------------- K7: branch 1 window attention (q ws=8, kv ds=4) --------------
__global__ void k_attn1(const h16* __restrict__ q, const h16* __restrict__ kd,
                        const h16* __restrict__ vd, h16* __restrict__ y) {
  __shared__ h16 Ql[64 * 48], Kl[16 * 48], Vc[32 * 48], Al[64 * 48], Yl[32 * 72];
  int wid = blockIdx.x;
  int b = wid >> 10, wy = (wid >> 5) & 31, wx = wid & 31;
  int tid = threadIdx.x;
  bool wrapx = (wx == 31);
  size_t base = (size_t)b * C_ * HW_;
  for (int sgi = tid; sgi < 256; sgi += 128) {       // Q (channels 32..63)
    int i = sgi & 7, c = sgi >> 3;
    int gy = (wy * 8 + i + 4) & 255;
    const h16* src = q + base + (size_t)(32 + c) * HW_ + gy * W_;
    int t0 = i * 8;
    if (!wrapx) {
      int gx0 = wx * 8 + 4;
      v4h a = *reinterpret_cast<const v4h*>(src + gx0);
      v4h bb = *reinterpret_cast<const v4h*>(src + gx0 + 4);
#pragma unroll
      for (int j = 0; j < 4; ++j) { Ql[(t0 + j) * 48 + c] = a[j]; Ql[(t0 + 4 + j) * 48 + c] = bb[j]; }
    } else {
#pragma unroll
      for (int j = 0; j < 8; ++j) Ql[(t0 + j) * 48 + c] = src[(wx * 8 + j + 4) & 255];
    }
  }
  for (int sgi = tid; sgi < 128; sgi += 128) {       // K rows of 4 tokens
    int i = sgi & 3, c = sgi >> 2;
    int gy = (wy * 4 + i + 2) & 127;
    const h16* src = kd + ((size_t)b * 32 + c) * (128 * 128) + gy * 128;
    int t0 = i * 4;
    if (!wrapx) {
      int gx0 = wx * 4 + 2;
      v2h a = *reinterpret_cast<const v2h*>(src + gx0);
      v2h bb = *reinterpret_cast<const v2h*>(src + gx0 + 2);
      Kl[t0 * 48 + c] = a[0]; Kl[(t0 + 1) * 48 + c] = a[1];
      Kl[(t0 + 2) * 48 + c] = bb[0]; Kl[(t0 + 3) * 48 + c] = bb[1];
    } else {
#pragma unroll
      for (int j = 0; j < 4; ++j) Kl[(t0 + j) * 48 + c] = src[(wx * 4 + j + 2) & 127];
    }
  }
  for (int e = tid; e < 512; e += 128) {             // zero K-dim pad of V
    int t = e & 15, c = e >> 4;
    Vc[c * 48 + 16 + t] = (h16)0.f;
  }
  if (!wrapx) {                                      // V: async b32 copies
    for (int sgi = tid; sgi < 256; sgi += 128) {
      int hf = sgi & 1, i = (sgi >> 1) & 3, c = sgi >> 3;
      int gy = (wy * 4 + i + 2) & 127, gx0 = wx * 4 + 2 + hf * 2;
      async_ld_b32((unsigned)(size_t)&Vc[c * 48 + i * 4 + hf * 2],
                   vd + ((size_t)b * 32 + c) * (128 * 128) + gy * 128 + gx0);
    }
  } else {
    for (int e = tid; e < 512; e += 128) {
      int t = e & 15, c = e >> 4;
      int gy = (wy * 4 + (t >> 2) + 2) & 127, gx = (wx * 4 + (t & 3) + 2) & 127;
      Vc[c * 48 + t] = vd[((size_t)b * 32 + c) * (128 * 128) + gy * 128 + gx];
    }
  }
  wait_async0();
  __syncthreads();
  int w = tid >> 5, l = tid & 31, row = l & 15;
  bool hi = l >= 16;
  int ab = hi ? 16 : 0;
  v16h aq = joinh(lds8(&Ql[(w * 16 + row) * 48 + ab]), lds8(&Ql[(w * 16 + row) * 48 + ab + 8]));
  int kb = hi ? 16 : 0;
  v16h bk = joinh(lds8(&Kl[row * 48 + kb]), lds8(&Kl[row * 48 + kb + 8]));
  v8f z0 = {0.f, 0.f, 0.f, 0.f, 0.f, 0.f, 0.f, 0.f};
  v8f lg = wmma16(aq, bk, z0);
#pragma unroll
  for (int r = 0; r < 8; ++r) lg[r] *= SCALE_;
#pragma unroll
  for (int r = 0; r < 8; ++r) {
    float mx = lg[r];
    for (int s = 1; s < 16; s <<= 1) mx = fmaxf(mx, __shfl_xor(mx, s, 32));
    float e = __expf(lg[r] - mx);
    float sm = e;
    for (int s = 1; s < 16; s <<= 1) sm += __shfl_xor(sm, s, 32);
    lg[r] = e / sm;
  }
#pragma unroll
  for (int r = 0; r < 8; ++r) {
    int qn = r + (hi ? 8 : 0);
    Al[(w * 16 + qn) * 48 + row] = (h16)lg[r];
    Al[(w * 16 + qn) * 48 + 16 + row] = (h16)0.f;
  }
  __syncthreads();
#pragma unroll
  for (int ct = 0; ct < 2; ++ct) {
    int ak = hi ? 16 : 0;
    v16h aa = joinh(lds8(&Al[(w * 16 + row) * 48 + ak]),
                    lds8(&Al[(w * 16 + row) * 48 + ak + 8]));
    int vk = hi ? 16 : 0;
    v16h bb = joinh(lds8(&Vc[(ct * 16 + row) * 48 + vk]),
                    lds8(&Vc[(ct * 16 + row) * 48 + vk + 8]));
    v8f o = wmma16(aa, bb, z0);
#pragma unroll
    for (int r = 0; r < 8; ++r)
      Yl[(ct * 16 + row) * 72 + w * 16 + r + (hi ? 8 : 0)] = (h16)o[r];
  }
  __syncthreads();
  for (int sgi = tid; sgi < 256; sgi += 128) {       // channels 32..63 out
    int i = sgi & 7, c = sgi >> 3;
    int gy = (wy * 8 + i + 4) & 255;
    h16* dg = y + base + (size_t)(32 + c) * HW_ + gy * W_;
    if (!wrapx) {
      int gx0 = wx * 8 + 4;
      *reinterpret_cast<v4h*>(dg + gx0)     = *reinterpret_cast<v4h*>(&Yl[c * 72 + i * 8]);
      *reinterpret_cast<v4h*>(dg + gx0 + 4) = *reinterpret_cast<v4h*>(&Yl[c * 72 + i * 8 + 4]);
    } else {
#pragma unroll
      for (int j = 0; j < 8; ++j) dg[(wx * 8 + j + 4) & 255] = Yl[c * 72 + i * 8 + j];
    }
  }
}

// ---------------- K8: spatial gate: z = y + sigmoid(sa.y + b) * x1 -------------
__global__ void k_sa(const h16* __restrict__ y, const h16* __restrict__ x1,
                     const float* __restrict__ saw, const float* __restrict__ sab,
                     h16* __restrict__ z) {
  int pix = blockIdx.x * 256 + threadIdx.x;
  int b = pix / HW_, m = pix - b * HW_;
  const h16* yp = y + (size_t)b * C_ * HW_ + m;
  const h16* xp = x1 + (size_t)b * C_ * HW_ + m;
  float dot = sab[0];
  for (int c = 0; c < C_; ++c) {
    __builtin_prefetch(xp + (size_t)c * HW_, 0, 0);   // warm x1 planes for pass 2
    dot += saw[c] * (float)yp[(size_t)c * HW_];
  }
  float s = 1.f / (1.f + __expf(-dot));
  h16* zp = z + (size_t)b * C_ * HW_ + m;
  for (int c = 0; c < C_; ++c)
    zp[(size_t)c * HW_] = (h16)((float)yp[(size_t)c * HW_] + s * (float)xp[(size_t)c * HW_]);
}

// ---------------- K9: final 1x1 conv via WMMA + bias + residual (f32 out) -------
__global__ void k_po(const h16* __restrict__ z, const float* __restrict__ pw,
                     const float* __restrict__ pb, const float* __restrict__ x0,
                     float* __restrict__ out) {
  __shared__ h16 Wl[C_ * C_];
  __shared__ h16 Xt[128 * 72];
  __shared__ float Otf[C_ * 128];
  int b = blockIdx.x / (HW_ / 128);
  int mb = (blockIdx.x % (HW_ / 128)) * 128;
  int tid = threadIdx.x;
  for (int i = tid; i < C_ * C_; i += 256) Wl[i] = (h16)pw[i];
  const h16* xp = z + (size_t)b * C_ * HW_ + mb;
  for (int e = tid * 4; e < 128 * C_; e += 256 * 4) {
    int c = e >> 7, m = e & 127;
    v4h val = *reinterpret_cast<const v4h*>(xp + (size_t)c * HW_ + m);
#pragma unroll
    for (int j = 0; j < 4; ++j) Xt[(m + j) * 72 + c] = val[j];
  }
  __syncthreads();
  int w = tid >> 5, l = tid & 31, row = l & 15;
  bool hi = l >= 16;
  int pcol = w * 16;
  v16h bfrag[2];
#pragma unroll
  for (int ks = 0; ks < 2; ++ks) {
    int kb = ks * 32 + (hi ? 16 : 0);
    bfrag[ks] = joinh(lds8(&Xt[(pcol + row) * 72 + kb]), lds8(&Xt[(pcol + row) * 72 + kb + 8]));
  }
#pragma unroll
  for (int rb = 0; rb < 4; ++rb) {
    v8f acc = {0.f, 0.f, 0.f, 0.f, 0.f, 0.f, 0.f, 0.f};
#pragma unroll
    for (int ks = 0; ks < 2; ++ks) {
      int kb = ks * 32 + (hi ? 16 : 0);
      v16h a = joinh(lds8(&Wl[(rb * 16 + row) * 64 + kb]),
                     lds8(&Wl[(rb * 16 + row) * 64 + kb + 8]));
      acc = wmma16(a, bfrag[ks], acc);
    }
#pragma unroll
    for (int r = 0; r < 8; ++r)
      Otf[(rb * 16 + r + (hi ? 8 : 0)) * 128 + pcol + row] = acc[r];
  }
  __syncthreads();
  for (int ch = tid; ch < 2048; ch += 256) {      // 64 o x 32 segs of 4 floats
    int o = ch >> 5, seg = ch & 31;
    size_t g = ((size_t)b * C_ + o) * HW_ + mb + seg * 4;
    v4f val = *reinterpret_cast<v4f*>(&Otf[o * 128 + seg * 4]);
    v4f xr = *reinterpret_cast<const v4f*>(x0 + g);
    float bias = pb[o];
    v4f rr;
#pragma unroll
    for (int j = 0; j < 4; ++j) rr[j] = val[j] + bias + xr[j];
    *reinterpret_cast<v4f*>(out + g) = rr;
  }
}

extern "C" void kernel_launch(void* const* d_in, const int* in_sizes, int n_in,
                              void* d_out, int out_size, void* d_ws, size_t ws_size,
                              hipStream_t stream) {
  const float* x0    = (const float*)d_in[0];
  const float* ln_w  = (const float*)d_in[1];
  const float* ln_b  = (const float*)d_in[2];
  const float* dw_w  = (const float*)d_in[3];
  const float* dw_b  = (const float*)d_in[4];
  const float* ew1   = (const float*)d_in[5];
  const float* ew2   = (const float*)d_in[6];
  const float* ew3   = (const float*)d_in[7];
  const float* ewc   = (const float*)d_in[8];
  const float* wq    = (const float*)d_in[9];
  const float* wk    = (const float*)d_in[10];
  const float* wv    = (const float*)d_in[11];
  const float* dn_w  = (const float*)d_in[12];
  const float* dn_b  = (const float*)d_in[13];
  const float* sa_w  = (const float*)d_in[14];
  const float* sa_b  = (const float*)d_in[15];
  const float* po_w  = (const float*)d_in[16];
  const float* po_b  = (const float*)d_in[17];
  float* out = (float*)d_out;

  const size_t NX = (size_t)B_ * C_ * HW_;
  char* ws = (char*)d_ws;
  size_t off = 0;
  auto carve = [&](size_t bytes) -> void* {
    void* p = ws + off; off += (bytes + 255) & ~(size_t)255; return p;
  };
  h16*   xh      = (h16*)carve(NX * 2);
  h16*   x1h     = (h16*)carve(NX * 2);
  h16*   qh      = (h16*)carve(NX * 2);
  h16*   kh      = (h16*)carve(NX * 2);
  h16*   vh      = (h16*)carve(NX * 2);
  h16*   yh      = (h16*)carve(NX * 2);
  h16*   kdn     = (h16*)carve((size_t)B_ * 32 * 128 * 128 * 2);
  h16*   vdn     = (h16*)carve((size_t)B_ * 32 * 128 * 128 * 2);
  float* partial = (float*)carve((size_t)B_ * C_ * 256 * 4);
  float* pool    = (float*)carve((size_t)B_ * C_ * 4);
  float* ca      = (float*)carve((size_t)B_ * C_ * 4);
  h16*   zh      = qh;  // reuse q buffer after attention

  k_ln  <<<(B_ * HW_) / 256, 256, 0, stream>>>(x0, ln_w, ln_b, xh);
  k_dw  <<<B_ * C_ * 256, 256, 0, stream>>>(xh, dw_w, dw_b, x1h, partial);
  k_pool<<<2, 256, 0, stream>>>(partial, pool);
  k_eca <<<B_, C_, 0, stream>>>(pool, ew1, ew2, ew3, ewc, ca);
  k_qkv <<<B_ * (HW_ / 128), 256, 0, stream>>>(xh, wq, wk, wv, ca, qh, kh, vh);
  k_down<<<(B_ * 32 * 128 * 128) / 256, 256, 0, stream>>>(kh, vh, dn_w, dn_b, kdn, vdn);
  k_attn0<<<B_ * 1024, 128, 0, stream>>>(qh, kh, vh, yh);
  k_attn1<<<B_ * 1024, 128, 0, stream>>>(qh, kdn, vdn, yh);
  k_sa  <<<(B_ * HW_) / 256, 256, 0, stream>>>(yh, x1h, sa_w, sa_b, zh);
  k_po  <<<B_ * (HW_ / 128), 256, 0, stream>>>(zh, po_w, po_b, x0, out);
}
